// CentroidDistance_74749610819752
// MI455X (gfx1250) — matrix operand order repaired and checked by reading, not compiled, not run on hardware
//
#include <hip/hip_runtime.h>
#include <hip/hip_bf16.h>

// Poincare centroid distance, MI455X (gfx1250, wave32, WMMA).
// d_in[0]: node_repr f32 [150000,256]
// d_in[1]: mask      f32 [150000,1]
// d_in[2]: centroid  f32 [256,256]
// d_out:   f32 [256] graph_centroid_dist  ++  f32 [150000*256] node_centroid_dist
// d_ws:    >=136KB scratch: [0]=mask_sum, [64..319]=sc, byte 2048..=f16 centroid copy

#define NNODES   150000
#define EMBED    256
#define NCENT    256
#define NSTRIPS  (NNODES / 16)          // 9375
#define WAVES_PB 8

typedef __attribute__((ext_vector_type(16))) _Float16 v16h;
typedef __attribute__((ext_vector_type(8)))  float    v8f;

__global__ __launch_bounds__(256) void cd_zero_init(float* gout, float* msum) {
    if (threadIdx.x == 0) msum[0] = 0.f;
    gout[threadIdx.x] = 0.f;
}

__global__ __launch_bounds__(256) void cd_mask_sum(const float* __restrict__ mask,
                                                   float* __restrict__ msum, int n) {
    __shared__ float red[256];
    float s = 0.f;
    for (int i = blockIdx.x * 256 + threadIdx.x; i < n; i += gridDim.x * 256)
        s += mask[i];
    red[threadIdx.x] = s;
    __syncthreads();
    for (int off = 128; off > 0; off >>= 1) {
        if (threadIdx.x < off) red[threadIdx.x] += red[threadIdx.x + off];
        __syncthreads();
    }
    if (threadIdx.x == 0) atomicAdd(msum, red[0]);
}

// f16 copy of centroid matrix + per-centroid squared norms (f32)
__global__ __launch_bounds__(256) void cd_prep_centroid(const float* __restrict__ w,
                                                        float* __restrict__ sc,
                                                        _Float16* __restrict__ wh) {
    const int c = threadIdx.x;           // one thread per centroid
    float s = 0.f;
    const float* row = w + (size_t)c * EMBED;
    _Float16*   orow = wh + (size_t)c * EMBED;
    #pragma unroll 8
    for (int k = 0; k < EMBED; ++k) {
        float v = row[k];
        s += v * v;
        orow[k] = (_Float16)v;
    }
    sc[c] = s;
}

__global__ __launch_bounds__(256) void cd_main(const float* __restrict__ x,
                                               const float* __restrict__ mask,
                                               const _Float16* __restrict__ wh,
                                               const float* __restrict__ sc,
                                               float* __restrict__ gout,
                                               float* __restrict__ outnode) {
    __shared__ float lacc[NCENT];
    const int tid = threadIdx.x;
    lacc[tid] = 0.f;
    __syncthreads();

    const int wave  = tid >> 5;
    const int lane  = tid & 31;
    const int strip = blockIdx.x * WAVES_PB + wave;

    if (strip < NSTRIPS) {               // wave-uniform: EXEC stays all-ones inside
        const int m    = lane & 15;      // this lane's A row within the strip
        const int h    = lane >> 4;      // half-wave selector (K-half / M-half)
        const int row0 = strip * 16;
        const float* xrow = x + (size_t)(row0 + m) * EMBED;

        // ---- Load A panel into WMMA fragment layout, f32->f16, track ||x||^2 ----
        v16h afrag[8];
        float ssq = 0.f;
        #pragma unroll
        for (int s = 0; s < 8; ++s) {
            const int kb = 32 * s;
            const float4* p0 = (const float4*)(xrow + kb + h * 16);       // K = kb+h*16 .. +7
            const float4* p1 = (const float4*)(xrow + kb + h * 16 + 8);   // K = kb+h*16+8 .. +15
            float4 f0 = p0[0], f1 = p0[1];
            float4 f2 = p1[0], f3 = p1[1];
            ssq += f0.x*f0.x + f0.y*f0.y + f0.z*f0.z + f0.w*f0.w
                 + f1.x*f1.x + f1.y*f1.y + f1.z*f1.z + f1.w*f1.w
                 + f2.x*f2.x + f2.y*f2.y + f2.z*f2.z + f2.w*f2.w
                 + f3.x*f3.x + f3.y*f3.y + f3.z*f3.z + f3.w*f3.w;
            v16h a;
            a[0]=(_Float16)f0.x; a[1]=(_Float16)f0.y; a[2]=(_Float16)f0.z; a[3]=(_Float16)f0.w;
            a[4]=(_Float16)f1.x; a[5]=(_Float16)f1.y; a[6]=(_Float16)f1.z; a[7]=(_Float16)f1.w;
            a[8]=(_Float16)f2.x; a[9]=(_Float16)f2.y; a[10]=(_Float16)f2.z; a[11]=(_Float16)f2.w;
            a[12]=(_Float16)f3.x; a[13]=(_Float16)f3.y; a[14]=(_Float16)f3.z; a[15]=(_Float16)f3.w;
            afrag[s] = a;
        }
        // lanes l and l+16 each hold half of row m's K range -> combine
        const float sx_m = ssq + __shfl_xor(ssq, 16, 32);
        const float mk_m = mask[row0 + m];

        // Epilogue needs sx/mask for rows M = 8*h + r (C/D layout), fetch via bpermute
        float sxM[8], mkM[8];
        #pragma unroll
        for (int r = 0; r < 8; ++r) {
            sxM[r] = __shfl(sx_m, h * 8 + r, 32);
            mkM[r] = __shfl(mk_m, h * 8 + r, 32);
        }

        // ---- 16 centroid tiles of 16, K=256 in 8 WMMA steps each ----
        for (int nt = 0; nt < 16; ++nt) {
            const int n0 = nt * 16 + m;                         // this lane's N column
            const _Float16* brow = wh + (size_t)n0 * EMBED + h * 16;
            v8f acc = {0.f, 0.f, 0.f, 0.f, 0.f, 0.f, 0.f, 0.f};
            #pragma unroll
            for (int s = 0; s < 8; ++s) {
                v16h b = *(const v16h*)(brow + 32 * s);         // contiguous 32B fragment
                acc = __builtin_amdgcn_wmma_f32_16x16x32_f16(
                        false, afrag[s], false, b, (short)0, acc, false, false);
            }

            const float scN = sc[n0];
            const float omc = 1.f - scN;
            float tsum = 0.f;
            #pragma unroll
            for (int r = 0; r < 8; ++r) {
                const float xc  = acc[r];
                const float sxv = sxM[r];
                float sq    = fmaxf(sxv + scN - 2.f * xc, 0.f);
                float denom = fmaxf((1.f - sxv) * omc, 1e-12f);
                float arg   = fmaxf(1.f + 2.f * sq / denom, 1.f + 1e-7f);
                float dist  = __logf(arg + __fsqrt_rn(__fmaf_rn(arg, arg, -1.f)));
                float val   = dist * mkM[r];
                outnode[(size_t)(row0 + 8 * h + r) * NCENT + n0] = val;
                tsum += val;
            }
            atomicAdd(&lacc[n0], tsum);    // ds_add_f32, 2-way lane conflict only
        }
    }

    __syncthreads();
    atomicAdd(&gout[tid], lacc[tid]);      // one global atomic per centroid per block
}

__global__ __launch_bounds__(256) void cd_final_scale(float* gout, const float* msum) {
    gout[threadIdx.x] *= (1.f / msum[0]);
}

extern "C" void kernel_launch(void* const* d_in, const int* in_sizes, int n_in,
                              void* d_out, int out_size, void* d_ws, size_t ws_size,
                              hipStream_t stream) {
    const float* x    = (const float*)d_in[0];
    const float* mask = (const float*)d_in[1];
    const float* w    = (const float*)d_in[2];

    float* out     = (float*)d_out;
    float* gout    = out;            // [256] graph_centroid_dist
    float* outnode = out + NCENT;    // [150000*256] node_centroid_dist

    float*    ws_f = (float*)d_ws;
    float*    msum = ws_f;                                // 1 f32
    float*    sc   = ws_f + 64;                           // 256 f32 @ byte 256
    _Float16* wh   = (_Float16*)((char*)d_ws + 2048);     // 256*256 f16 @ byte 2048

    cd_zero_init<<<1, 256, 0, stream>>>(gout, msum);
    cd_mask_sum<<<120, 256, 0, stream>>>(mask, msum, NNODES);
    cd_prep_centroid<<<1, 256, 0, stream>>>(w, sc, wh);

    const int blocks = (NSTRIPS + WAVES_PB - 1) / WAVES_PB;   // 1172
    cd_main<<<blocks, 256, 0, stream>>>(x, mask, wh, sc, gout, outnode);

    cd_final_scale<<<1, 256, 0, stream>>>(gout, msum);
}
